// UFOAttention_31035433680935
// MI455X (gfx1250) — compile-verified, matrix operand order checked by reading, not run on hardware
//
#include <hip/hip_runtime.h>
#include <hip/hip_bf16.h>

// ---------------------------------------------------------------------------
// UFO attention for MI455X (gfx1250, wave32).
// All GEMMs run through v_wmma_f32_16x16x32_bf16 with fp32 accumulation.
// Intermediates are bf16 to halve HBM traffic (the binding resource here).
// A-tiles are staged with gfx1250 async copies (global_load_async_to_lds_b128,
// ASYNCcnt) and double-buffered to overlap HBM latency with WMMA issue.
// ---------------------------------------------------------------------------

typedef __attribute__((ext_vector_type(16))) __bf16 v16bf;
typedef __attribute__((ext_vector_type(8)))  float  v8f;

__device__ __forceinline__ v8f wmma_bf16(v16bf a, v16bf b, v8f c) {
  // 8 args: (neg_a, A, neg_b, B, c_mod, C, reuse_a, reuse_b)
  return __builtin_amdgcn_wmma_f32_16x16x32_bf16(false, a, false, b, (short)0, c,
                                                 false, false);
}

// 16-bit A/B fragment layout (ISA 7.12.2): lane L holds row/col (L%16), with
// K elements { kg*8 .. kg*8+7 } and { 16+kg*8 .. 16+kg*8+7 }, kg = L/16.
// `rowp` points at the 32-element K-run for this lane's row/col (16B aligned).
__device__ __forceinline__ v16bf load_frag(const __bf16* rowp, int kg) {
  union { v16bf v; uint4 u[2]; } f;
  f.u[0] = *(const uint4*)(rowp + kg * 8);
  f.u[1] = *(const uint4*)(rowp + 16 + kg * 8);
  return f.v;
}

// Async global->LDS copy of 16 bytes per lane (tracked by ASYNCcnt).
// lds_off is the workgroup-relative LDS byte address (low 32 bits of the
// generic pointer to a __shared__ object).
__device__ __forceinline__ void async_copy_b128(unsigned lds_off,
                                                const __bf16* gptr) {
  asm volatile("global_load_async_to_lds_b128 %0, %1, off"
               :: "v"(lds_off), "v"(gptr)
               : "memory");
}

// LDS row pitch (elements) for 32-wide K tiles; multiple of 16 keeps 32B align.
#define KPITCH 48

// ---------------------------------------------------------------------------
// Generic bf16 GEMM: C[M,N] = A[M,K] * B[K,N] + bias.  A,B row-major bf16.
// Block tile 128x128, K-step 32.  8 waves: 4(M) x 2(N), wave tile 32x64.
// A tile: double-buffered async copies.  B tile: VGPR transpose into LDS.
// M,N,K must be multiples of 128/128/32 (true for all uses here).
// ---------------------------------------------------------------------------
template <bool OUT_BF16>
__global__ __launch_bounds__(256) void gemm_bf16_kernel(
    const __bf16* __restrict__ A, const __bf16* __restrict__ B,
    const float* __restrict__ bias, void* __restrict__ C,
    int M, int N, int K) {
  __shared__ alignas(16) __bf16 As[2][128 * KPITCH];  // [buf][m][k]
  __shared__ alignas(16) __bf16 Bs[128 * KPITCH];     // transposed: [n][k]
  (void)M;

  const int tid  = threadIdx.x;
  const int lane = tid & 31;
  const int w    = tid >> 5;
  const int lm   = lane & 15;
  const int kg   = lane >> 4;
  const int wm   = w >> 1;  // 0..3
  const int wn   = w & 1;   // 0..1
  const int n0   = blockIdx.x * 128;
  const int m0   = blockIdx.y * 128;

  const v8f vzero = {0.f, 0.f, 0.f, 0.f, 0.f, 0.f, 0.f, 0.f};
  v8f acc[2][4];
#pragma unroll
  for (int i = 0; i < 2; ++i)
#pragma unroll
    for (int j = 0; j < 4; ++j) acc[i][j] = vzero;

  // Per-thread A staging slot: row m, 16 contiguous bf16 at col ko (two b128s)
  const int am = tid >> 1;
  const int ko = (tid & 1) * 16;
  const unsigned ldsA0 = (unsigned)(uintptr_t)&As[0][am * KPITCH + ko];
  const unsigned ldsA1 = (unsigned)(uintptr_t)&As[1][am * KPITCH + ko];
  const __bf16* gA = A + (size_t)(m0 + am) * K + ko;

  // Per-thread B staging slot
  const int kk = tid >> 4;        // 0..15
  const int nn = (tid & 15) * 8;  // 0..120

#define ISSUE_A(kbv, lds0)                 \
  do {                                     \
    async_copy_b128((lds0), gA + (kbv));   \
    async_copy_b128((lds0) + 16, gA + (kbv) + 8); \
  } while (0)

  const int steps = K >> 5;
  ISSUE_A(0, ldsA0);

  for (int it = 0; it < steps; ++it) {
    const int kb = it << 5;
    // --- fetch B tile 32x128 into regs (overlaps outstanding async A) ---
    union { uint4 u; __bf16 h[8]; } ld0, ld1;
    ld0.u = *(const uint4*)(B + (size_t)(kb + kk) * N + n0 + nn);
    ld1.u = *(const uint4*)(B + (size_t)(kb + kk + 16) * N + n0 + nn);
    // --- prefetch next A tile into the other buffer ---
    if (it + 1 < steps) ISSUE_A(kb + 32, (it & 1) ? ldsA0 : ldsA1);
    // --- transpose-scatter B into Bs[n][k] ---
#pragma unroll
    for (int j = 0; j < 8; ++j) {
      Bs[(nn + j) * KPITCH + kk]      = ld0.h[j];
      Bs[(nn + j) * KPITCH + kk + 16] = ld1.h[j];
    }
    // Current A tile's 2 async ops were issued first; async loads complete
    // in order, so <=2 outstanding leaves only the prefetched tile in flight.
    if (it + 1 < steps) asm volatile("s_wait_asynccnt 2" ::: "memory");
    else                asm volatile("s_wait_asynccnt 0" ::: "memory");
    __syncthreads();

    const __bf16* Acur = As[it & 1];
    v16bf af[2], bf[4];
#pragma unroll
    for (int mi = 0; mi < 2; ++mi)
      af[mi] = load_frag(&Acur[(wm * 32 + mi * 16 + lm) * KPITCH], kg);
#pragma unroll
    for (int ni = 0; ni < 4; ++ni)
      bf[ni] = load_frag(&Bs[(wn * 64 + ni * 16 + lm) * KPITCH], kg);
#pragma unroll
    for (int mi = 0; mi < 2; ++mi)
#pragma unroll
      for (int ni = 0; ni < 4; ++ni)
        acc[mi][ni] = wmma_bf16(af[mi], bf[ni], acc[mi][ni]);
    __syncthreads();
  }
#undef ISSUE_A

  // Epilogue: C/D layout — VGPR r: lanes 0-15 M=r, lanes 16-31 M=r+8; N=lane%16
#pragma unroll
  for (int mi = 0; mi < 2; ++mi)
#pragma unroll
    for (int ni = 0; ni < 4; ++ni)
#pragma unroll
      for (int r = 0; r < 8; ++r) {
        const int m = m0 + wm * 32 + mi * 16 + r + 8 * kg;
        const int n = n0 + wn * 64 + ni * 16 + lm;
        const float val = acc[mi][ni][r] + bias[n];
        if (OUT_BF16) ((__bf16*)C)[(size_t)m * N + n] = (__bf16)val;
        else          ((float*)C)[(size_t)m * N + n]  = val;
      }
}

// ---------------------------------------------------------------------------
// kv[b,h,dk,dv] = sum_n k[b,n,h,dk] * v[b,n,h,dv]; split-K over n with fp32
// global atomics.  qkv layout: [b*8192+n][1536] with k at +512, v at +1024.
// grid = (64 bh, 8 chunks); chunk = 1024 tokens.  Both operands need a
// transpose, so staging stays on the VGPR path.
// ---------------------------------------------------------------------------
__global__ __launch_bounds__(256) void kv_gemm_kernel(
    const __bf16* __restrict__ qkv, float* __restrict__ kv) {
  __shared__ alignas(16) __bf16 kT[64 * KPITCH];  // [dk][n_local]
  __shared__ alignas(16) __bf16 vT[64 * KPITCH];  // [dv][n_local]

  const int tid  = threadIdx.x;
  const int lane = tid & 31;
  const int w    = tid >> 5;
  const int lm   = lane & 15;
  const int kg   = lane >> 4;
  const int bh   = blockIdx.x;  // 0..63
  const int b    = bh >> 3;
  const int h    = bh & 7;
  const int mi   = w >> 1;       // dk tile 0..3
  const int nib  = (w & 1) * 2;  // dv tiles {0,1} or {2,3}

  const v8f vzero = {0.f, 0.f, 0.f, 0.f, 0.f, 0.f, 0.f, 0.f};
  v8f acc0 = vzero, acc1 = vzero;

  const int nl  = tid >> 3;       // 0..31 token within step
  const int dko = (tid & 7) * 8;  // 0..56

  const int nbeg = blockIdx.y * 1024;
  for (int nb = nbeg; nb < nbeg + 1024; nb += 32) {
    const size_t rowbase = (size_t)(b * 8192 + nb + nl) * 1536 + h * 64 + dko;
    union { uint4 u; __bf16 e[8]; } lk, lv;
    lk.u = *(const uint4*)(qkv + rowbase + 512);
    lv.u = *(const uint4*)(qkv + rowbase + 1024);
#pragma unroll
    for (int j = 0; j < 8; ++j) {
      kT[(dko + j) * KPITCH + nl] = lk.e[j];
      vT[(dko + j) * KPITCH + nl] = lv.e[j];
    }
    __syncthreads();
    v16bf a  = load_frag(&kT[(mi * 16 + lm) * KPITCH], kg);
    v16bf b0 = load_frag(&vT[(nib * 16 + lm) * KPITCH], kg);
    v16bf b1 = load_frag(&vT[((nib + 1) * 16 + lm) * KPITCH], kg);
    acc0 = wmma_bf16(a, b0, acc0);
    acc1 = wmma_bf16(a, b1, acc1);
    __syncthreads();
  }

  float* dst = kv + (size_t)bh * 4096;
#pragma unroll
  for (int r = 0; r < 8; ++r) {
    const int m = mi * 16 + r + 8 * kg;
    atomicAdd(&dst[m * 64 + nib * 16 + lm], acc0[r]);
    atomicAdd(&dst[m * 64 + (nib + 1) * 16 + lm], acc1[r]);
  }
}

// ---------------------------------------------------------------------------
// kv_n: normalize each (b,h,dk) row over dv, scale by gamma[h], write bf16
// TRANSPOSED as kvnT[b,h,dv,dk] (so it is directly a WMMA B operand).
// One wave per row; 4096 rows. grid = 512 blocks x 256.
// ---------------------------------------------------------------------------
__global__ __launch_bounds__(256) void kv_norm_kernel(
    const float* __restrict__ kv, const float* __restrict__ gamma,
    __bf16* __restrict__ kvnT) {
  const int row  = blockIdx.x * 8 + (threadIdx.x >> 5);  // 0..4095
  const int lane = threadIdx.x & 31;
  const int bh = row >> 6, dk = row & 63, h = (row >> 6) & 7;
  const float f0 = kv[(size_t)row * 64 + lane];
  const float f1 = kv[(size_t)row * 64 + lane + 32];
  float s = f0 * f0 + f1 * f1;
#pragma unroll
  for (int off = 16; off > 0; off >>= 1) s += __shfl_xor(s, off, 32);
  const float scale = gamma[h] * rsqrtf(s);
  kvnT[((size_t)bh * 64 + lane) * 64 + dk]      = (__bf16)(f0 * scale);
  kvnT[((size_t)bh * 64 + lane + 32) * 64 + dk] = (__bf16)(f1 * scale);
}

// ---------------------------------------------------------------------------
// q_n: normalize each (b,n,h) row over dk, scale by gamma[h].
// qkv cols [0,512) hold q. Output qn[token][h*64+dk] bf16.
// One wave per row; 524288 rows. grid = 65536 blocks x 256.
// ---------------------------------------------------------------------------
__global__ __launch_bounds__(256) void q_norm_kernel(
    const __bf16* __restrict__ qkv, const float* __restrict__ gamma,
    __bf16* __restrict__ qn) {
  const int row  = blockIdx.x * 8 + (threadIdx.x >> 5);
  const int lane = threadIdx.x & 31;
  const int token = row >> 3, h = row & 7;
  const size_t base = (size_t)token * 1536 + h * 64;
  const float f0 = (float)qkv[base + lane];
  const float f1 = (float)qkv[base + lane + 32];
  float s = f0 * f0 + f1 * f1;
#pragma unroll
  for (int off = 16; off > 0; off >>= 1) s += __shfl_xor(s, off, 32);
  const float scale = gamma[h] * rsqrtf(s);
  const size_t ob = (size_t)token * 512 + h * 64;
  qn[ob + lane]      = (__bf16)(f0 * scale);
  qn[ob + lane + 32] = (__bf16)(f1 * scale);
}

// ---------------------------------------------------------------------------
// attn[token][h*64+dv] = qn[token, h, :] @ kvnT[b,h]^T.  K=64: fragments load
// straight from global (kv tile is 8KB, lives in L2/WGP$).  No LDS.
// grid = (64 token-chunks of 128, 64 bh); wave handles 16 tokens x 64 dv.
// ---------------------------------------------------------------------------
__global__ __launch_bounds__(256) void attn_gemm_kernel(
    const __bf16* __restrict__ qn, const __bf16* __restrict__ kvnT,
    __bf16* __restrict__ attn) {
  const int tid  = threadIdx.x;
  const int lane = tid & 31;
  const int w    = tid >> 5;
  const int lm   = lane & 15;
  const int kg   = lane >> 4;
  const int bh = blockIdx.y, b = bh >> 3, h = bh & 7;
  const int tok0 = blockIdx.x * 128 + w * 16;

  const v8f vzero = {0.f, 0.f, 0.f, 0.f, 0.f, 0.f, 0.f, 0.f};
  v8f acc[4] = {vzero, vzero, vzero, vzero};

  const __bf16* arow  = qn + (size_t)(b * 8192 + tok0 + lm) * 512 + h * 64;
  const __bf16* bbase = kvnT + (size_t)bh * 4096;

#pragma unroll
  for (int kb = 0; kb < 64; kb += 32) {
    v16bf a = load_frag(arow + kb, kg);
#pragma unroll
    for (int ni = 0; ni < 4; ++ni) {
      v16bf bf = load_frag(bbase + (size_t)(ni * 16 + lm) * 64 + kb, kg);
      acc[ni] = wmma_bf16(a, bf, acc[ni]);
    }
  }

#pragma unroll
  for (int ni = 0; ni < 4; ++ni)
#pragma unroll
    for (int r = 0; r < 8; ++r) {
      const int g = b * 8192 + tok0 + r + 8 * kg;
      attn[(size_t)g * 512 + h * 64 + ni * 16 + lm] = (__bf16)acc[ni][r];
    }
}

// ---------------------------------------------------------------------------
// prep: fp32->bf16 conversions, weight packing [Wq|Wk|Wv], bias packing,
// zero the kv accumulator (atomics accumulate into it each launch).
// ---------------------------------------------------------------------------
__global__ __launch_bounds__(256) void prep_kernel(
    const float* __restrict__ queries, const float* __restrict__ Wq,
    const float* __restrict__ bq, const float* __restrict__ Wk,
    const float* __restrict__ bk, const float* __restrict__ Wv,
    const float* __restrict__ bv, const float* __restrict__ Wo,
    __bf16* __restrict__ q_bf, __bf16* __restrict__ wqkv,
    __bf16* __restrict__ wo_bf, float* __restrict__ biasq,
    float* __restrict__ kvz) {
  const size_t i = (size_t)blockIdx.x * 256 + threadIdx.x;
  if (i < 33554432ull) q_bf[i] = (__bf16)queries[i];
  if (i < 786432ull) {
    const int k = (int)(i / 1536), j = (int)(i % 1536);
    const float v = (j < 512)    ? Wq[k * 512 + j]
                    : (j < 1024) ? Wk[k * 512 + j - 512]
                                 : Wv[k * 512 + j - 1024];
    wqkv[i] = (__bf16)v;
  }
  if (i < 262144ull) { wo_bf[i] = (__bf16)Wo[i]; kvz[i] = 0.f; }
  if (i < 1536ull) {
    const int j = (int)i;
    biasq[j] = (j < 512) ? bq[j] : (j < 1024) ? bk[j - 512] : bv[j - 1024];
  }
}

// ---------------------------------------------------------------------------
extern "C" void kernel_launch(void* const* d_in, const int* in_sizes, int n_in,
                              void* d_out, int out_size, void* d_ws,
                              size_t ws_size, hipStream_t stream) {
  (void)in_sizes; (void)n_in; (void)out_size; (void)ws_size;

  const float* queries = (const float*)d_in[0];
  const float* Wq      = (const float*)d_in[1];
  const float* bq      = (const float*)d_in[2];
  const float* Wk      = (const float*)d_in[3];
  const float* bk      = (const float*)d_in[4];
  const float* Wv      = (const float*)d_in[5];
  const float* bv      = (const float*)d_in[6];
  const float* Wo      = (const float*)d_in[7];
  const float* bo      = (const float*)d_in[8];
  const float* gamma   = (const float*)d_in[9];

  char* ws = (char*)d_ws;
  __bf16* q_bf  = (__bf16*)(ws);              //  67,108,864 B
  __bf16* wqkv  = (__bf16*)(ws + 67108864);   //   1,572,864 B
  __bf16* wo_bf = (__bf16*)(ws + 68681728);   //     524,288 B
  float*  biasq = (float*) (ws + 69206016);   //       6,144 B (pad 8K)
  float*  kv    = (float*) (ws + 69214208);   //   1,048,576 B
  __bf16* kvnT  = (__bf16*)(ws + 70262784);   //     524,288 B
  __bf16* qkv   = (__bf16*)(ws + 70787072);   // 201,326,592 B
  __bf16* qn    = (__bf16*)(ws + 272113664);  //  67,108,864 B
  __bf16* attn  = (__bf16*)(ws + 339222528);  //  67,108,864 B (~406 MB)

  // 1) conversions / packing / kv zero-init
  prep_kernel<<<131072, 256, 0, stream>>>(queries, Wq, bq, Wk, bk, Wv, bv, Wo,
                                          q_bf, wqkv, wo_bf, biasq, kv);
  // 2) fused QKV projection: [65536x512] x [512x1536] -> bf16 qkv
  gemm_bf16_kernel<true><<<dim3(12, 512), 256, 0, stream>>>(
      q_bf, wqkv, biasq, (void*)qkv, 65536, 1536, 512);
  // 3) kv = K^T V per (b,h), split-K atomics
  kv_gemm_kernel<<<dim3(64, 8), 256, 0, stream>>>(qkv, kv);
  // 4) normalize kv (over dv) -> bf16 transposed; normalize q (over dk)
  kv_norm_kernel<<<512, 256, 0, stream>>>(kv, gamma, kvnT);
  q_norm_kernel<<<65536, 256, 0, stream>>>(qkv, gamma, qn);
  // 5) attn = Qn @ KVn per (b,h)
  attn_gemm_kernel<<<dim3(64, 64), 256, 0, stream>>>(qn, kvnT, attn);
  // 6) final projection: [65536x512] x [512x512] + bo -> fp32 out
  gemm_bf16_kernel<false><<<dim3(4, 512), 256, 0, stream>>>(
      attn, wo_bf, bo, d_out, 65536, 512, 512);
}
